// LoRALinear4bit_15814069584451
// MI455X (gfx1250) — matrix operand order repaired and checked by reading, not compiled, not run on hardware
//
#include <hip/hip_runtime.h>

// ---------------------------------------------------------------------------
// LoRA 4-bit linear for MI455X (gfx1250, wave32, WMMA).
// y = x @ dequant_nf4(W)^T + 2 * (x @ A) @ B
// M=8192 (B*S), K=IN=4096, N=OUT=4096.
// Strategy: dequant NF4 -> bf16 W once (L2-resident), convert x -> bf16 once
// (fused with the tiny LoRA t = x@A reduction), then a tiled
// v_wmma_f32_16x16x32_bf16 GEMM with double-buffered LDS staged via
// GLOBAL_LOAD_ASYNC_TO_LDS_B128 (ASYNCcnt). LoRA is folded in as one extra
// K=32 WMMA step using zero-padded bf16 t and scaled B^T.
// ---------------------------------------------------------------------------

typedef __bf16 bf16;
typedef bf16  v16bf __attribute__((ext_vector_type(16)));
typedef bf16  v8bf  __attribute__((ext_vector_type(8)));
typedef float v8f   __attribute__((ext_vector_type(8)));

union U16 { v8bf h[2]; v16bf v; };

__device__ __constant__ float NF4c[16] = {
    -1.0f, -0.6961928009986877f, -0.5250730514526367f, -0.39491748809814453f,
    -0.28444138169288635f, -0.18477343022823334f, -0.09105003625154495f, 0.0f,
    0.07958029955625534f, 0.16093020141124725f, 0.24611230194568634f,
    0.33791524171829224f, 0.44070982933044434f, 0.5626170039176941f,
    0.6989699602127075f, 1.0f};

static __device__ __forceinline__ bf16 f2bf(float f) { return (bf16)f; }

// Shapes (fixed by the reference)
#define MM 8192
#define KK 4096
#define NN 4096

// Async global -> LDS copy (gfx1250, tracked by ASYNCcnt). Per-lane:
// LDS[%0] = MEM[%1], 16 bytes. "off" = no SADDR.
#define ASYNC_LD_B128(ldsoff, gaddr)                                  \
    asm volatile("global_load_async_to_lds_b128 %0, %1, off"          \
                 :: "v"(ldsoff), "v"(gaddr))
#define WAIT_ASYNC0() asm volatile("s_wait_asynccnt 0x0" ::: "memory")

// ---------------------------------------------------------------------------
// Kernel 1: x fp32 -> bf16, fused with t = x @ lora_A (M x 16), written as
// bf16 zero-padded to M x 32 so the LoRA path becomes one WMMA K-step.
// One block = 8 rows of x; wave w handles row w (8 waves of 32 lanes).
// lora_A is staged through LDS in 512-row chunks.
// ---------------------------------------------------------------------------
__global__ void __launch_bounds__(256)
prep_x(const float* __restrict__ x, const float* __restrict__ lora_A,
       bf16* __restrict__ Xb, bf16* __restrict__ Tpad)
{
    __shared__ float As[512 * 16];   // 32 KiB chunk of lora_A
    __shared__ float tacc[8][16];

    const int tid  = threadIdx.x;
    const int lane = tid & 31;
    const int w    = tid >> 5;                 // wave id == local row
    const int m    = blockIdx.x * 8 + w;

    if (tid < 128) ((float*)tacc)[tid] = 0.0f;

    float tloc[16];
#pragma unroll
    for (int r = 0; r < 16; ++r) tloc[r] = 0.0f;

    for (int kc = 0; kc < KK; kc += 512) {
        __syncthreads();                       // protect As reuse (+ tacc init)
#pragma unroll
        for (int i = 0; i < 8; ++i) {
            int idx = tid + i * 256;           // float4 index, 2048 total
            ((float4*)As)[idx] = ((const float4*)(lora_A + (size_t)kc * 16))[idx];
        }
        __syncthreads();

        for (int it = 0; it < 16; ++it) {
            int kl = it * 32 + lane;           // 0..511 within chunk
            float xv = x[(size_t)m * KK + kc + kl];
            Xb[(size_t)m * KK + kc + kl] = f2bf(xv);
            const float4* ap = (const float4*)(As + kl * 16);
#pragma unroll
            for (int q = 0; q < 4; ++q) {
                float4 a4 = ap[q];
                tloc[q * 4 + 0] += xv * a4.x;
                tloc[q * 4 + 1] += xv * a4.y;
                tloc[q * 4 + 2] += xv * a4.z;
                tloc[q * 4 + 3] += xv * a4.w;
            }
        }
    }
    __syncthreads();
#pragma unroll
    for (int r = 0; r < 16; ++r) atomicAdd(&tacc[w][r], tloc[r]);  // ds_add_f32
    __syncthreads();

    // 8 rows x 32 cols of Tpad (cols 16..31 are zero padding)
    {
        int row = tid >> 5, c = tid & 31;
        float v = (c < 16) ? tacc[row][c] : 0.0f;
        Tpad[(size_t)(blockIdx.x * 8 + row) * 32 + c] = f2bf(v);
    }
}

// ---------------------------------------------------------------------------
// Kernel 2: NF4 dequant -> bf16 W [OUT, IN] row-major. 8 elems/thread.
// ---------------------------------------------------------------------------
__global__ void __launch_bounds__(256)
dequant_w(const int* __restrict__ q, const float* __restrict__ absmax,
          bf16* __restrict__ Wb)
{
    const size_t base = ((size_t)blockIdx.x * 256 + threadIdx.x) * 8;
    const int o = (int)(base >> 12);           // IN = 4096
    const int i = (int)(base & 4095);
    const float s = absmax[(size_t)o * (KK / 64) + (i >> 6)];

    const int4* qp = (const int4*)(q + base);
    int4 q0 = qp[0];
    int4 q1 = qp[1];

    union { bf16 h[8]; uint4 u; } wv;
    wv.h[0] = f2bf(NF4c[q0.x & 15] * s);
    wv.h[1] = f2bf(NF4c[q0.y & 15] * s);
    wv.h[2] = f2bf(NF4c[q0.z & 15] * s);
    wv.h[3] = f2bf(NF4c[q0.w & 15] * s);
    wv.h[4] = f2bf(NF4c[q1.x & 15] * s);
    wv.h[5] = f2bf(NF4c[q1.y & 15] * s);
    wv.h[6] = f2bf(NF4c[q1.z & 15] * s);
    wv.h[7] = f2bf(NF4c[q1.w & 15] * s);
    *(uint4*)(Wb + base) = wv.u;
}

// ---------------------------------------------------------------------------
// Kernel 3: BpT[n][k] = 2 * lora_B[k][n] for k<16 (bf16), 0 for k=16..31.
// ---------------------------------------------------------------------------
__global__ void __launch_bounds__(256)
make_bpt(const float* __restrict__ lora_B, bf16* __restrict__ BpT)
{
    int id = blockIdx.x * 256 + threadIdx.x;   // 0 .. 4096*32-1
    int n = id >> 5, k = id & 31;
    float v = (k < 16) ? (2.0f * lora_B[(size_t)k * NN + n]) : 0.0f;
    BpT[id] = f2bf(v);
}

// ---------------------------------------------------------------------------
// Kernel 4: tiled bf16 WMMA GEMM, out = Xb @ Wb^T + Tpad @ BpT^T.
// Block tile 128x256, 8 waves in 2x4, 64x64 per wave (16 WMMA accumulators).
// K-step 32, double-buffered LDS staged with async global->LDS copies;
// 112-byte row pitch (16B aligned, conflict friendly for ds_load_b128).
// ---------------------------------------------------------------------------
#define ABYTES (128 * 112)              // 14336
#define BBYTES (256 * 112)              // 28672
#define BUFB   (ABYTES + BBYTES)        // 43008 per buffer, x2 = 86016

__global__ void __launch_bounds__(256)
gemm_wmma(const bf16* __restrict__ Xb, const bf16* __restrict__ Wb,
          const bf16* __restrict__ Tpad, const bf16* __restrict__ BpT,
          float* __restrict__ out)
{
    extern __shared__ unsigned char lds[];

    const int tid  = threadIdx.x;
    const int lane = tid & 31;
    const int wid  = tid >> 5;
    const int wm   = wid >> 2;          // 0..1
    const int wn   = wid & 3;           // 0..3
    const int m0   = blockIdx.y * 128;
    const int n0   = blockIdx.x * 256;
    const int l    = lane & 15;
    const int ahalf = (lane < 16) ? 0 : 16;   // byte offset of K-chunk 0 (A frag)
    const int bhalf = (lane < 16) ? 0 : 16;   // byte offset of K-chunk 0 (B frag)

    v8f acc[4][4];
    {
        v8f z = {0.f, 0.f, 0.f, 0.f, 0.f, 0.f, 0.f, 0.f};
#pragma unroll
        for (int i = 0; i < 4; ++i)
#pragma unroll
            for (int j = 0; j < 4; ++j) acc[i][j] = z;
    }

    // staging-chunk coordinates (16B chunks: A 512, B 1024; 256 threads)
    // chunk c: row = c>>2, col = (c&3)*8 elements (16B)
    const unsigned ldsbase = (unsigned)(size_t)(void*)&lds[0];
    unsigned laA[2], laB[4];            // LDS byte offsets (buffer 0)
    unsigned long long gaA[2], gaB[4];  // global byte addresses (current K-tile)
#pragma unroll
    for (int i = 0; i < 2; ++i) {
        int c = tid + i * 256, row = c >> 2, col = (c & 3) * 8;
        laA[i] = ldsbase + (unsigned)(row * 112 + col * 2);
        gaA[i] = (unsigned long long)(Xb + (size_t)(m0 + row) * KK + col);
    }
#pragma unroll
    for (int i = 0; i < 4; ++i) {
        int c = tid + i * 256, row = c >> 2, col = (c & 3) * 8;
        laB[i] = ldsbase + (unsigned)(ABYTES + row * 112 + col * 2);
        gaB[i] = (unsigned long long)(Wb + (size_t)(n0 + row) * KK + col);
    }

    // prologue: async-stage K-tile 0 into buffer 0
#pragma unroll
    for (int i = 0; i < 2; ++i) ASYNC_LD_B128(laA[i], gaA[i]);
#pragma unroll
    for (int i = 0; i < 4; ++i) ASYNC_LD_B128(laB[i], gaB[i]);
    WAIT_ASYNC0();
    __syncthreads();

    const int nk = KK / 32;
    for (int kt = 0; kt < nk; ++kt) {
        const int  cur  = kt & 1;
        const bool more = (kt + 1) < nk;

        if (more) {                     // async-stage next K-tile (64B step)
            const unsigned boff = (unsigned)((cur ^ 1) * BUFB);
#pragma unroll
            for (int i = 0; i < 2; ++i) {
                gaA[i] += 64;
                ASYNC_LD_B128(laA[i] + boff, gaA[i]);
            }
#pragma unroll
            for (int i = 0; i < 4; ++i) {
                gaB[i] += 64;
                ASYNC_LD_B128(laB[i] + boff, gaB[i]);
            }
        }

        // compute on current buffer
        unsigned char* A  = lds + cur * BUFB;
        unsigned char* Bm = A + ABYTES;
        v16bf af[4], bfr[4];
#pragma unroll
        for (int i = 0; i < 4; ++i) {
            const unsigned char* p = A + (wm * 64 + i * 16 + l) * 112 + ahalf;
            U16 u;
            u.h[0] = *(const v8bf*)p;                 // K 0-7 / 8-15
            u.h[1] = *(const v8bf*)(p + 32);          // K 16-23 / 24-31
            af[i] = u.v;
        }
#pragma unroll
        for (int j = 0; j < 4; ++j) {
            const unsigned char* p = Bm + (wn * 64 + j * 16 + l) * 112 + bhalf;
            U16 u;
            u.h[0] = *(const v8bf*)p;                 // K 0-7 / 8-15
            u.h[1] = *(const v8bf*)(p + 32);          // K 16-23 / 24-31
            bfr[j] = u.v;
        }
#pragma unroll
        for (int i = 0; i < 4; ++i)
#pragma unroll
            for (int j = 0; j < 4; ++j)
                acc[i][j] = __builtin_amdgcn_wmma_f32_16x16x32_bf16(
                    false, af[i], false, bfr[j], (short)0, acc[i][j], false, false);

        WAIT_ASYNC0();                  // next tile fully in LDS
        __syncthreads();
    }

    // LoRA epilogue: one extra K=32 WMMA step from Tpad (M x 32) and BpT (N x 32)
    {
        v16bf af[4], bfr[4];
#pragma unroll
        for (int i = 0; i < 4; ++i) {
            const unsigned char* p =
                (const unsigned char*)Tpad + (size_t)(m0 + wm * 64 + i * 16 + l) * 64 + ahalf;
            U16 u;
            u.h[0] = *(const v8bf*)p;
            u.h[1] = *(const v8bf*)(p + 32);
            af[i] = u.v;
        }
#pragma unroll
        for (int j = 0; j < 4; ++j) {
            const unsigned char* p =
                (const unsigned char*)BpT + (size_t)(n0 + wn * 64 + j * 16 + l) * 64 + bhalf;
            U16 u;
            u.h[0] = *(const v8bf*)p;
            u.h[1] = *(const v8bf*)(p + 32);
            bfr[j] = u.v;
        }
#pragma unroll
        for (int i = 0; i < 4; ++i)
#pragma unroll
            for (int j = 0; j < 4; ++j)
                acc[i][j] = __builtin_amdgcn_wmma_f32_16x16x32_bf16(
                    false, af[i], false, bfr[j], (short)0, acc[i][j], false, false);
    }

    // store: element (VGPR r, lane) = (M = r + 8*(lane>=16), N = lane&15)
#pragma unroll
    for (int i = 0; i < 4; ++i)
#pragma unroll
        for (int j = 0; j < 4; ++j) {
            const int mbase = m0 + wm * 64 + i * 16 + ((lane >> 4) * 8);
            const int nn    = n0 + wn * 64 + j * 16 + l;
#pragma unroll
            for (int r = 0; r < 8; ++r)
                out[(size_t)(mbase + r) * NN + nn] = acc[i][j][r];
        }
}

// ---------------------------------------------------------------------------
extern "C" void kernel_launch(void* const* d_in, const int* in_sizes, int n_in,
                              void* d_out, int out_size, void* d_ws, size_t ws_size,
                              hipStream_t stream)
{
    const float* x      = (const float*)d_in[0];
    const int*   q_idx  = (const int*)d_in[1];
    const float* absmax = (const float*)d_in[2];
    const float* lora_A = (const float*)d_in[3];
    const float* lora_B = (const float*)d_in[4];
    float*       out    = (float*)d_out;

    char* ws = (char*)d_ws;
    bf16* Xb   = (bf16*)(ws);                                   // 64 MiB
    bf16* Wb   = (bf16*)(ws + (size_t)67108864);                // 32 MiB
    bf16* Tpad = (bf16*)(ws + (size_t)67108864 + 33554432);     // 0.5 MiB
    bf16* BpT  = (bf16*)(ws + (size_t)67108864 + 33554432 + 524288); // 0.25 MiB

    prep_x  <<<MM / 8, 256, 0, stream>>>(x, lora_A, Xb, Tpad);
    dequant_w<<<(size_t)NN * KK / 2048, 256, 0, stream>>>(q_idx, absmax, Wb);
    make_bpt<<<NN * 32 / 256, 256, 0, stream>>>(lora_B, BpT);

    dim3 grid(NN / 256, MM / 128);
    gemm_wmma<<<grid, 256, 2 * BUFB, stream>>>(Xb, Wb, Tpad, BpT, out);
}
